// LSTM_74466142978402
// MI455X (gfx1250) — matrix operand order repaired and checked by reading, not compiled, not run on hardware
//
#include <hip/hip_runtime.h>
#include <hip/hip_bf16.h>

#define B_ 256
#define T_ 512
#define I_ 256
#define H_ 1024
#define MB 2   // batch tiles per wave: halves gate-weight L2 replication

typedef __attribute__((ext_vector_type(16))) __bf16 v16bf;
typedef __attribute__((ext_vector_type(8)))  __bf16 v8bf;
typedef __attribute__((ext_vector_type(8)))  float  v8f;

// ---------------------------------------------------------------------------
// fp32 -> bf16 hi/lo split (3xBF16 fp32-emulation operands)
// ---------------------------------------------------------------------------
__global__ __launch_bounds__(256) void split_bf16(const float* __restrict__ src,
                                                  __bf16* __restrict__ hi,
                                                  __bf16* __restrict__ lo, int n) {
    for (int i = blockIdx.x * blockDim.x + threadIdx.x; i < n;
         i += gridDim.x * blockDim.x) {
        float x = src[i];
        __bf16 h = (__bf16)x;
        hi[i] = h;
        lo[i] = (__bf16)(x - (float)h);
    }
}

__global__ __launch_bounds__(256) void init_state(float* __restrict__ c,
                                                  __bf16* __restrict__ h_hi,
                                                  __bf16* __restrict__ h_lo, int n) {
    int i = blockIdx.x * blockDim.x + threadIdx.x;
    if (i < n) {
        c[i] = 0.0f;
        h_hi[i] = (__bf16)0.0f;
        h_lo[i] = (__bf16)0.0f;
    }
}

// ---------------------------------------------------------------------------
// A-operand loader: 16-bit A 16x32, contiguous k order (same as B loads):
//   lane L: m = L&15, kh = L>>4
//   elems 0..7  -> K = kk + kh*16 + i       (8 contiguous bf16 = 16B)
//   elems 8..15 -> K = kk + kh*16 + 8 + i   (8 contiguous bf16 = 16B)
// ---------------------------------------------------------------------------
__device__ __forceinline__ v16bf load_a(const __bf16* __restrict__ row_kk, int kh) {
    v8bf a0 = *(const v8bf*)(row_kk + kh * 16);
    v8bf a1 = *(const v8bf*)(row_kk + kh * 16 + 8);
    v16bf a;
#pragma unroll
    for (int i = 0; i < 8; ++i) { a[i] = a0[i]; a[i + 8] = a1[i]; }
    return a;
}

__device__ __forceinline__ float sigmoid_f(float x) {
    return 1.0f / (1.0f + __expf(-x));
}

// ---------------------------------------------------------------------------
// One LSTM timestep. Grid = (H/16, B/(16*MB)), block = 32 (one wave).
// Each wave: MB 16x16 output tiles for all 4 gates (B-operand reused across
// batch tiles), K = I_ (input projection) + H_ (recurrent).
// 3xBF16 fp32 emulation: acc += Ahi*Bhi + Ahi*Blo + Alo*Bhi.
// ---------------------------------------------------------------------------
__global__ __launch_bounds__(32) void lstm_step(
    const __bf16* __restrict__ ts_hi, const __bf16* __restrict__ ts_lo,
    const __bf16* __restrict__ wih_hi, const __bf16* __restrict__ wih_lo,
    const __bf16* __restrict__ whh_hi, const __bf16* __restrict__ whh_lo,
    const float* __restrict__ b_ih, const float* __restrict__ b_hh,
    const __bf16* __restrict__ hin_hi, const __bf16* __restrict__ hin_lo,
    __bf16* __restrict__ hout_hi, __bf16* __restrict__ hout_lo,
    float* __restrict__ c, int t) {
    const int lane = threadIdx.x;          // 0..31
    const int n0 = blockIdx.x * 16;        // hidden-tile base
    const int b0 = blockIdx.y * 16 * MB;   // batch-tile base (MB tiles)
    const int nl = lane & 15;
    const int kh = lane >> 4;

    // Accumulators seeded with combined bias (depends only on column n).
    v8f acc[4][MB];
#pragma unroll
    for (int g = 0; g < 4; ++g) {
        float bv = b_ih[g * H_ + n0 + nl] + b_hh[g * H_ + n0 + nl];
#pragma unroll
        for (int mb = 0; mb < MB; ++mb)
#pragma unroll
            for (int r = 0; r < 8; ++r) acc[g][mb][r] = bv;
    }

    // ---- input projection contribution: K = I_ ----
    {
        const __bf16* arow_hi[MB];
        const __bf16* arow_lo[MB];
#pragma unroll
        for (int mb = 0; mb < MB; ++mb) {
            arow_hi[mb] = ts_hi + ((size_t)(b0 + mb * 16 + nl) * T_ + t) * I_;
            arow_lo[mb] = ts_lo + ((size_t)(b0 + mb * 16 + nl) * T_ + t) * I_;
        }
        for (int kk = 0; kk < I_; kk += 32) {
            v16bf ahi[MB], alo[MB];
#pragma unroll
            for (int mb = 0; mb < MB; ++mb) {
                ahi[mb] = load_a(arow_hi[mb] + kk, kh);
                alo[mb] = load_a(arow_lo[mb] + kk, kh);
            }
#pragma unroll
            for (int g = 0; g < 4; ++g) {
                size_t boff = (size_t)(g * H_ + n0 + nl) * I_ + kk + kh * 16;
                v16bf bhi = *(const v16bf*)(wih_hi + boff);
                v16bf blo = *(const v16bf*)(wih_lo + boff);
#pragma unroll
                for (int mb = 0; mb < MB; ++mb) {
                    v8f a = acc[g][mb];
                    a = __builtin_amdgcn_wmma_f32_16x16x32_bf16(false, ahi[mb], false, bhi, (short)0, a, false, false);
                    a = __builtin_amdgcn_wmma_f32_16x16x32_bf16(false, ahi[mb], false, blo, (short)0, a, false, false);
                    a = __builtin_amdgcn_wmma_f32_16x16x32_bf16(false, alo[mb], false, bhi, (short)0, a, false, false);
                    acc[g][mb] = a;
                }
            }
        }
    }

    // ---- recurrent contribution: K = H_ ----
    {
        const __bf16* arow_hi[MB];
        const __bf16* arow_lo[MB];
#pragma unroll
        for (int mb = 0; mb < MB; ++mb) {
            arow_hi[mb] = hin_hi + (size_t)(b0 + mb * 16 + nl) * H_;
            arow_lo[mb] = hin_lo + (size_t)(b0 + mb * 16 + nl) * H_;
        }
        for (int kk = 0; kk < H_; kk += 32) {
            v16bf ahi[MB], alo[MB];
#pragma unroll
            for (int mb = 0; mb < MB; ++mb) {
                ahi[mb] = load_a(arow_hi[mb] + kk, kh);
                alo[mb] = load_a(arow_lo[mb] + kk, kh);
            }
#pragma unroll
            for (int g = 0; g < 4; ++g) {
                size_t boff = (size_t)(g * H_ + n0 + nl) * H_ + kk + kh * 16;
                v16bf bhi = *(const v16bf*)(whh_hi + boff);
                v16bf blo = *(const v16bf*)(whh_lo + boff);
#pragma unroll
                for (int mb = 0; mb < MB; ++mb) {
                    v8f a = acc[g][mb];
                    a = __builtin_amdgcn_wmma_f32_16x16x32_bf16(false, ahi[mb], false, bhi, (short)0, a, false, false);
                    a = __builtin_amdgcn_wmma_f32_16x16x32_bf16(false, ahi[mb], false, blo, (short)0, a, false, false);
                    a = __builtin_amdgcn_wmma_f32_16x16x32_bf16(false, alo[mb], false, bhi, (short)0, a, false, false);
                    acc[g][mb] = a;
                }
            }
        }
    }

    // ---- fused gate nonlinearities + state update ----
    // C/D layout: VGPR r, lane L -> m = (L>>4)*8 + r, n = L&15
#pragma unroll
    for (int mb = 0; mb < MB; ++mb) {
#pragma unroll
        for (int r = 0; r < 8; ++r) {
            int m = b0 + mb * 16 + kh * 8 + r;
            int n = n0 + nl;
            size_t idx = (size_t)m * H_ + n;
            float iv = sigmoid_f(acc[0][mb][r]);
            float fv = sigmoid_f(acc[1][mb][r]);
            float gv = tanhf(acc[2][mb][r]);
            float ov = sigmoid_f(acc[3][mb][r]);
            float cn = fv * c[idx] + iv * gv;
            c[idx] = cn;
            float hn = ov * tanhf(cn);
            __bf16 hh = (__bf16)hn;
            hout_hi[idx] = hh;
            hout_lo[idx] = (__bf16)(hn - (float)hh);
        }
    }
}

// ---------------------------------------------------------------------------
// out[b] = dot(c[b,:], fc_w) + fc_b
// ---------------------------------------------------------------------------
__global__ __launch_bounds__(256) void fc_out(const float* __restrict__ c,
                                              const float* __restrict__ fc_w,
                                              const float* __restrict__ fc_b,
                                              float* __restrict__ out) {
    __shared__ float red[256];
    int b = blockIdx.x;
    int tid = threadIdx.x;
    float s = 0.0f;
    for (int j = tid; j < H_; j += 256) s += c[(size_t)b * H_ + j] * fc_w[j];
    red[tid] = s;
    __syncthreads();
    for (int off = 128; off > 0; off >>= 1) {
        if (tid < off) red[tid] += red[tid + off];
        __syncthreads();
    }
    if (tid == 0) out[b] = red[0] + fc_b[0];
}

// ---------------------------------------------------------------------------
extern "C" void kernel_launch(void* const* d_in, const int* in_sizes, int n_in,
                              void* d_out, int out_size, void* d_ws, size_t ws_size,
                              hipStream_t stream) {
    const float* ts   = (const float*)d_in[0];  // [B,T,I]
    const float* w_ih = (const float*)d_in[1];  // [4H,I]
    const float* w_hh = (const float*)d_in[2];  // [4H,H]
    const float* b_ih = (const float*)d_in[3];  // [4H]
    const float* b_hh = (const float*)d_in[4];  // [4H]
    const float* fc_w = (const float*)d_in[5];  // [1,H]
    const float* fc_b = (const float*)d_in[6];  // [1]
    float* out = (float*)d_out;                 // [B,1] fp32

    // Workspace carving (all section sizes are multiples of 256B).
    char* ws = (char*)d_ws;
    size_t off = 0;
    const size_t n_ts  = (size_t)B_ * T_ * I_;   // 33.55M
    const size_t n_wih = (size_t)4 * H_ * I_;    // 1.05M
    const size_t n_whh = (size_t)4 * H_ * H_;    // 4.19M
    const size_t n_st  = (size_t)B_ * H_;        // 262144

    __bf16* ts_hi  = (__bf16*)(ws + off); off += n_ts  * 2;
    __bf16* ts_lo  = (__bf16*)(ws + off); off += n_ts  * 2;
    __bf16* wih_hi = (__bf16*)(ws + off); off += n_wih * 2;
    __bf16* wih_lo = (__bf16*)(ws + off); off += n_wih * 2;
    __bf16* whh_hi = (__bf16*)(ws + off); off += n_whh * 2;
    __bf16* whh_lo = (__bf16*)(ws + off); off += n_whh * 2;
    __bf16* h_hi[2];
    __bf16* h_lo[2];
    h_hi[0] = (__bf16*)(ws + off); off += n_st * 2;
    h_lo[0] = (__bf16*)(ws + off); off += n_st * 2;
    h_hi[1] = (__bf16*)(ws + off); off += n_st * 2;
    h_lo[1] = (__bf16*)(ws + off); off += n_st * 2;
    float* c = (float*)(ws + off); off += n_st * 4;

    // One-shot precision splits (HBM-bound, ~290 MB total traffic).
    split_bf16<<<4096, 256, 0, stream>>>(ts, ts_hi, ts_lo, (int)n_ts);
    split_bf16<<<1024, 256, 0, stream>>>(w_ih, wih_hi, wih_lo, (int)n_wih);
    split_bf16<<<2048, 256, 0, stream>>>(w_hh, whh_hi, whh_lo, (int)n_whh);
    init_state<<<(int)(n_st / 256), 256, 0, stream>>>(c, h_hi[0], h_lo[0], (int)n_st);

    // Sequential recurrence: ping-pong h buffers across launches.
    dim3 grid(H_ / 16, B_ / (16 * MB));  // 64 x 8 waves
    for (int t = 0; t < T_; ++t) {
        int p = t & 1;
        lstm_step<<<grid, 32, 0, stream>>>(ts_hi, ts_lo, wih_hi, wih_lo,
                                           whh_hi, whh_lo, b_ih, b_hh,
                                           h_hi[p], h_lo[p],
                                           h_hi[1 - p], h_lo[1 - p], c, t);
    }

    fc_out<<<B_, 256, 0, stream>>>(c, fc_w, fc_b, out);
}
